// ProsodyPredictorWrapper_40767829574459
// MI455X (gfx1250) — matrix operand order repaired and checked by reading, not compiled, not run on hardware
//
#include <hip/hip_runtime.h>
#include <cstdint>

#define DEVINL __device__ __forceinline__

typedef __attribute__((ext_vector_type(4)))  unsigned int v4u;
typedef __attribute__((ext_vector_type(8)))  float        v8f;
typedef __attribute__((ext_vector_type(16))) __bf16       v16bf;

struct Frag2 { v4u lo, hi; };

DEVINL v16bf fragcast(Frag2 f) { return __builtin_bit_cast(v16bf, f); }

DEVINL unsigned short f2bf(float x) {
  unsigned int u = __float_as_uint(x);
  u += 0x7fffu + ((u >> 16) & 1u);             // round-to-nearest-even
  return (unsigned short)(u >> 16);
}
DEVINL float bf2f(unsigned short h) { return __uint_as_float(((unsigned int)h) << 16); }
DEVINL float sigm(float x) { return 1.0f / (1.0f + __expf(-x)); }

// A fragment (16x32 bf16): lane m = lane%16; elements 0..7 -> K = k0..k0+7,
// elements 8..15 -> K = k0+16..k0+23, where k0 = ks*32 + (lane/16)*8.
DEVINL v16bf ldA(const unsigned short* rowbase, int k0) {
  Frag2 f;
  f.lo = *reinterpret_cast<const v4u*>(rowbase + k0);
  f.hi = *reinterpret_cast<const v4u*>(rowbase + k0 + 16);
  return fragcast(f);
}
DEVINL Frag2 ldraw(const unsigned short* rowbase, int k0) {
  Frag2 f;
  f.lo = *reinterpret_cast<const v4u*>(rowbase + k0);
  f.hi = *reinterpret_cast<const v4u*>(rowbase + k0 + 16);
  return f;
}
// B fragment (32x16 bf16) from [o][k] storage: lane o fixed, K = bp..bp+7 and bp+16..bp+23.
DEVINL v16bf ldB(const unsigned short* bp) {
  Frag2 f;
  f.lo = *reinterpret_cast<const v4u*>(bp);
  f.hi = *reinterpret_cast<const v4u*>(bp + 16);
  return fragcast(f);
}
DEVINL v8f wmma_bf16(v16bf a, v16bf b, v8f c) {
  return __builtin_amdgcn_wmma_f32_16x16x32_bf16(false, a, false, b, (short)0, c, false, false);
}
DEVINL v8f vzero() { v8f z = {0.f,0.f,0.f,0.f,0.f,0.f,0.f,0.f}; return z; }

// -------- gfx1250 async load-to-LDS (ASYNCcnt path) ------------------------
#if defined(__has_builtin)
#if __has_builtin(__builtin_amdgcn_s_wait_asynccnt)
#define WAIT_ASYNC(N) __builtin_amdgcn_s_wait_asynccnt(N)
#endif
#endif
#ifndef WAIT_ASYNC
#define WAIT_ASYNC(N) asm volatile("s_wait_asynccnt %0" ::"i"(N) : "memory")
#endif

// Per-lane async copy of 16 bytes global->LDS (GLOBAL_LOAD_ASYNC_TO_LDS_B128).
DEVINL void async_g2l_b128(unsigned lds_byte_addr, const void* gaddr) {
  asm volatile("global_load_async_to_lds_b128 %0, %1, off"
               :: "v"(lds_byte_addr), "v"((unsigned long long)(uintptr_t)gaddr)
               : "memory");
}

// Stage a 64(rows) x 32(bf16) B tile into LDS: 256 x 16B chunks, 128 threads.
DEVINL void stage_b_tile(const unsigned short* Bsrc, int Ksz, int koff,
                         unsigned short* dstbuf, int tid) {
#pragma unroll
  for (int i = 0; i < 2; ++i) {
    const int c = tid + i * 128;
    const int op = c >> 2, kc = c & 3;
    const unsigned short* g = Bsrc + (size_t)op * Ksz + koff + kc * 8;
    const unsigned lo = (unsigned)(uintptr_t)(dstbuf + op * 32 + kc * 8);
    async_g2l_b128(lo, g);
  }
}

// ---------------------------------------------------------------------------
// Sizes: N=32, T=512 (padded rows 516), V=256, C=H=512, H2=256, E=256, P=64
// ---------------------------------------------------------------------------

__global__ __launch_bounds__(256) void embed_kernel(
    const int* __restrict__ ids, const float* __restrict__ emb,
    const float* __restrict__ spkr, unsigned short* __restrict__ act) {
  const int bid = blockIdx.x;         // n*516 + tp
  const int n = bid / 516, tp = bid % 516, t = tp - 2;
  const int c = threadIdx.x;
  const size_t base = ((size_t)n * 516 + tp) * 512;
  if (t < 0 || t >= 512) { act[base + c] = 0; act[base + c + 256] = 0; return; }
  const int id = ids[n * 512 + t];
  const float mk = spkr[n * 512 + t];
  act[base + c]       = f2bf(emb[(size_t)id * 512 + c] * mk);
  act[base + c + 256] = f2bf(emb[(size_t)id * 512 + c + 256] * mk);
}

__global__ __launch_bounds__(256) void padzero_kernel(unsigned short* __restrict__ act) {
  const int bid = blockIdx.x;           // n*4 + pi
  const int n = bid >> 2, pi = bid & 3;
  const int tp = (pi < 2) ? pi : 512 + pi;
  const size_t base = ((size_t)n * 516 + tp) * 512 + threadIdx.x;
  act[base] = 0; act[base + 256] = 0;
}

// conv_w (3,O=512,C=512,K=5) -> bf16 [l][k][o][c]
__global__ __launch_bounds__(256) void convw_prepack(
    const float* __restrict__ src, unsigned short* __restrict__ dst) {
  const int idx = blockIdx.x * 256 + threadIdx.x;
  if (idx >= 3 * 5 * 512 * 512) return;
  int c = idx & 511;
  int r = idx >> 9;
  int o = r & 511; r >>= 9;
  int k = r % 5, lyr = r / 5;
  dst[idx] = f2bf(src[(((size_t)lyr * 512 + o) * 512 + c) * 5 + k]);
}

__global__ __launch_bounds__(256) void cast_bf16(
    const float* __restrict__ src, unsigned short* __restrict__ dst,
    int rows, int src_stride, int src_off, int cols) {
  const int idx = blockIdx.x * 256 + threadIdx.x;
  if (idx >= rows * cols) return;
  const int r = idx / cols, c = idx % cols;
  dst[idx] = f2bf(src[(size_t)r * src_stride + src_off + c]);
}

// Conv layer as 5 shifted GEMMs + fused bias/BN/ReLU/mask epilogue.
// Block = 128 threads (4 waves) -> 128(M) x 64(N); wave = 32(M) x 64(N).
// B tile async-staged in LDS (double buffered, shared by all waves and both
// m-subtiles); A register-pipelined; K loop unrolled by 2 (no reg rotation).
__global__ __launch_bounds__(128) void conv_gemm_kernel(
    const unsigned short* __restrict__ Ain, const unsigned short* __restrict__ Wl,
    const float* __restrict__ cb, const float* __restrict__ bnm,
    const float* __restrict__ bng, const float* __restrict__ bnv,
    const float* __restrict__ bnb, const float* __restrict__ spkr,
    unsigned short* __restrict__ Aout) {
  __shared__ __align__(16) unsigned short bstage[2][64 * 32];
  const int tid = threadIdx.x;
  const int w = tid >> 5, l = tid & 31, gl = l >> 4, ln = l & 15;
  const int m0 = blockIdx.x * 128 + w * 32;
  const int ob = blockIdx.y * 64;
  const int rM0 = m0 + ln,  n0 = rM0 >> 9, t0 = rM0 & 511;
  const int rM1 = rM0 + 16, n1 = rM1 >> 9, t1 = rM1 & 511;
  v8f acc[2][4];
#pragma unroll
  for (int mi = 0; mi < 2; ++mi)
#pragma unroll
    for (int nt = 0; nt < 4; ++nt) acc[mi][nt] = vzero();

  stage_b_tile(Wl + (size_t)ob * 512, 512, 0, &bstage[0][0], tid);
  Frag2 af[2][2];
  af[0][0] = ldraw(Ain + ((size_t)n0 * 516 + t0) * 512, gl * 8);
  af[0][1] = ldraw(Ain + ((size_t)n1 * 516 + t1) * 512, gl * 8);
  af[1][0] = af[0][0]; af[1][1] = af[0][1];

  auto step = [&](int s, int pb, int nb) {
    const int sn = s + 1;
    if (sn < 80) {
      const int k2 = sn >> 4, ks2 = sn & 15;
      stage_b_tile(Wl + (size_t)k2 * 512 * 512 + (size_t)ob * 512, 512, ks2 * 32,
                   &bstage[nb][0], tid);
      af[nb][0] = ldraw(Ain + ((size_t)n0 * 516 + t0 + k2) * 512, ks2 * 32 + gl * 8);
      af[nb][1] = ldraw(Ain + ((size_t)n1 * 516 + t1 + k2) * 512, ks2 * 32 + gl * 8);
      WAIT_ASYNC(2);                     // B(s) complete (B(s+1) in flight)
    } else {
      WAIT_ASYNC(0);
    }
    __syncthreads();                     // B(s) visible to all waves
    const v16bf av0 = fragcast(af[pb][0]);
    const v16bf av1 = fragcast(af[pb][1]);
    const unsigned short* bb = &bstage[pb][gl * 8];
#pragma unroll
    for (int nt = 0; nt < 4; ++nt) {
      const v16bf bv = ldB(bb + (nt * 16 + ln) * 32);
      acc[0][nt] = wmma_bf16(av0, bv, acc[0][nt]);
      acc[1][nt] = wmma_bf16(av1, bv, acc[1][nt]);
    }
    __syncthreads();                     // done reading buf before overwrite
  };
  for (int s = 0; s < 80; s += 2) { step(s, 0, 1); step(s + 1, 1, 0); }

#pragma unroll
  for (int mi = 0; mi < 2; ++mi)
#pragma unroll
    for (int nt = 0; nt < 4; ++nt) {
      const int o = ob + nt * 16 + ln;
      const float sc = bng[o] * rsqrtf(bnv[o] + 1e-5f);
      const float c0 = cb[o] - bnm[o];
      const float b0 = bnb[o];
#pragma unroll
      for (int r = 0; r < 8; ++r) {
        const int rr = m0 + mi * 16 + gl * 8 + r;
        const int n2 = rr >> 9, t2 = rr & 511;
        float y = (acc[mi][nt][r] + c0) * sc + b0;
        y = fmaxf(y, 0.f) * spkr[n2 * 512 + t2];
        Aout[((size_t)n2 * 516 + t2 + 2) * 512 + o] = f2bf(y);
      }
    }
}

// Generic bf16 GEMM (K = 512), out scattered as [t][n][NO] bf16.
// Same structure as conv_gemm: 128x64 block tile, 32x64 per wave.
__global__ __launch_bounds__(128) void gemm_bf16_kernel(
    const unsigned short* __restrict__ A, int a_n_stride, int a_t_off,
    const unsigned short* __restrict__ B,
    unsigned short* __restrict__ out, int NO) {
  __shared__ __align__(16) unsigned short bstage[2][64 * 32];
  const int tid = threadIdx.x;
  const int w = tid >> 5, l = tid & 31, gl = l >> 4, ln = l & 15;
  const int m0 = blockIdx.x * 128 + w * 32;
  const int ob = blockIdx.y * 64;
  const int rM0 = m0 + ln,  n0 = rM0 >> 9, t0 = rM0 & 511;
  const int rM1 = rM0 + 16, n1 = rM1 >> 9, t1 = rM1 & 511;
  const unsigned short* ar0 = A + ((size_t)n0 * a_n_stride + t0 + a_t_off) * 512;
  const unsigned short* ar1 = A + ((size_t)n1 * a_n_stride + t1 + a_t_off) * 512;
  const unsigned short* Bb = B + (size_t)ob * 512;
  v8f acc[2][4];
#pragma unroll
  for (int mi = 0; mi < 2; ++mi)
#pragma unroll
    for (int nt = 0; nt < 4; ++nt) acc[mi][nt] = vzero();

  stage_b_tile(Bb, 512, 0, &bstage[0][0], tid);
  Frag2 af[2][2];
  af[0][0] = ldraw(ar0, gl * 8);
  af[0][1] = ldraw(ar1, gl * 8);
  af[1][0] = af[0][0]; af[1][1] = af[0][1];

  auto step = [&](int s, int pb, int nb) {
    const int sn = s + 1;
    if (sn < 16) {
      stage_b_tile(Bb, 512, sn * 32, &bstage[nb][0], tid);
      af[nb][0] = ldraw(ar0, sn * 32 + gl * 8);
      af[nb][1] = ldraw(ar1, sn * 32 + gl * 8);
      WAIT_ASYNC(2);
    } else {
      WAIT_ASYNC(0);
    }
    __syncthreads();
    const v16bf av0 = fragcast(af[pb][0]);
    const v16bf av1 = fragcast(af[pb][1]);
    const unsigned short* bb = &bstage[pb][gl * 8];
#pragma unroll
    for (int nt = 0; nt < 4; ++nt) {
      const v16bf bv = ldB(bb + (nt * 16 + ln) * 32);
      acc[0][nt] = wmma_bf16(av0, bv, acc[0][nt]);
      acc[1][nt] = wmma_bf16(av1, bv, acc[1][nt]);
    }
    __syncthreads();
  };
  for (int s = 0; s < 16; s += 2) { step(s, 0, 1); step(s + 1, 1, 0); }

#pragma unroll
  for (int mi = 0; mi < 2; ++mi)
#pragma unroll
    for (int nt = 0; nt < 4; ++nt) {
      const int o = ob + nt * 16 + ln;
#pragma unroll
      for (int r = 0; r < 8; ++r) {
        const int rr = m0 + mi * 16 + gl * 8 + r;
        const int n2 = rr >> 9, t2 = rr & 511;
        out[((size_t)t2 * 32 + n2) * NO + o] = f2bf(acc[mi][nt][r]);
      }
    }
}

// LSTM recurrence, one direction. Persistent single workgroup, 16 waves.
__global__ __launch_bounds__(512) void lstm_kernel(
    const unsigned short* __restrict__ gx, const unsigned short* __restrict__ whh,
    const float* __restrict__ bih, const float* __restrict__ bhh,
    float* __restrict__ text, int reverse, int colOff) {
  __shared__ __align__(16) float c_s[32 * 256];
  __shared__ __align__(16) unsigned short hbf[32 * 256];
  const int tid = threadIdx.x;
  const int w = tid >> 5, l = tid & 31, gl = l >> 4, ln = l & 15;
  for (int i = tid; i < 32 * 256; i += 512) { c_s[i] = 0.f; hbf[i] = 0; }
  __syncthreads();

  for (int s = 0; s < 512; ++s) {
    const int t = reverse ? (511 - s) : s;
    if (s + 1 < 512) {                    // prefetch next gate block (64KB)
      const int t2 = reverse ? (510 - s) : (s + 1);
      __builtin_prefetch((const char*)(gx + ((size_t)t2 * 32) * 1024) + tid * 128, 0, 1);
    }
    v8f acc[2][4];
#pragma unroll
    for (int pp = 0; pp < 2; ++pp)
#pragma unroll
      for (int g4 = 0; g4 < 4; ++g4) acc[pp][g4] = vzero();

    const unsigned short* ar0 = &hbf[ln * 256];
    const unsigned short* ar1 = &hbf[(16 + ln) * 256];
    for (int ks = 0; ks < 8; ++ks) {
      const int k0 = ks * 32 + gl * 8;
      const v16bf av0 = ldA(ar0, k0);
      const v16bf av1 = ldA(ar1, k0);
#pragma unroll
      for (int g4 = 0; g4 < 4; ++g4) {
        const v16bf bv = ldB(whh + (size_t)(g4 * 256 + w * 16 + ln) * 256 + ks * 32 + gl * 8);
        acc[0][g4] = wmma_bf16(av0, bv, acc[0][g4]);
        acc[1][g4] = wmma_bf16(av1, bv, acc[1][g4]);
      }
    }
    __syncthreads();                       // all reads of hbf done
#pragma unroll
    for (int pp = 0; pp < 2; ++pp) {
      const int j = w * 16 + ln;
      const float bs0 = bih[j] + bhh[j];
      const float bs1 = bih[256 + j] + bhh[256 + j];
      const float bs2 = bih[512 + j] + bhh[512 + j];
      const float bs3 = bih[768 + j] + bhh[768 + j];
#pragma unroll
      for (int r = 0; r < 8; ++r) {
        const int Mrow = pp * 16 + gl * 8 + r;           // batch index n
        const unsigned short* gp = gx + ((size_t)t * 32 + Mrow) * 1024 + j;
        const float ip = acc[pp][0][r] + bf2f(gp[0]) + bs0;
        const float fp = acc[pp][1][r] + bf2f(gp[256]) + bs1;
        const float gv = acc[pp][2][r] + bf2f(gp[512]) + bs2;
        const float op = acc[pp][3][r] + bf2f(gp[768]) + bs3;
        const float co = c_s[Mrow * 256 + j];
        const float cn = sigm(fp) * co + sigm(ip) * tanhf(gv);
        const float hv = sigm(op) * tanhf(cn);
        c_s[Mrow * 256 + j] = cn;
        hbf[Mrow * 256 + j] = f2bf(hv);
        text[(((size_t)Mrow * 512) + t) * 512 + colOff + j] = hv;
      }
    }
    __syncthreads();                       // new h visible before next step
  }
}

__global__ __launch_bounds__(256) void sp_kernel(
    const float* __restrict__ speed, const float* __restrict__ w1,
    const float* __restrict__ b1, const float* __restrict__ w2,
    const float* __restrict__ b2, float* __restrict__ spb) {
  const int n = blockIdx.x >> 1;
  const int c = ((blockIdx.x & 1) << 8) + threadIdx.x;
  const float s = speed[n];
  float acc = 0.f;
  for (int e = 0; e < 256; ++e) {
    const float a = fmaxf(s * w1[e] + b1[e], 0.f);
    acc += a * w2[(size_t)c * 256 + e];
  }
  spb[n * 512 + c] = tanhf(acc + b2[c]);
}

__global__ __launch_bounds__(256) void add_style_kernel(
    float* __restrict__ text, const float* __restrict__ style,
    const float* __restrict__ spb, unsigned short* __restrict__ encbf) {
  const int bid = blockIdx.x;            // n*512 + t
  const int n = bid >> 9;
#pragma unroll
  for (int kk = 0; kk < 2; ++kk) {
    const int c = threadIdx.x + (kk << 8);
    const size_t off = (size_t)bid * 512 + c;
    const float v = text[off] + style[off] + spb[n * 512 + c];
    text[off] = v;
    encbf[off] = f2bf(v);
  }
}

// GRU recurrence + projection feedback, persistent single workgroup.
__global__ __launch_bounds__(512) void gru_kernel(
    const unsigned short* __restrict__ gix, const unsigned short* __restrict__ whh,
    const unsigned short* __restrict__ wp, const unsigned short* __restrict__ pw,
    const float* __restrict__ bih, const float* __restrict__ bhh,
    const float* __restrict__ pb, float* __restrict__ outs) {
  __shared__ __align__(16) float h_s[32 * 256];
  __shared__ __align__(16) unsigned short hbf[32 * 256];
  __shared__ __align__(16) unsigned short pbf[32 * 64];
  const int tid = threadIdx.x;
  const int w = tid >> 5, l = tid & 31, gl = l >> 4, ln = l & 15;
  for (int i = tid; i < 32 * 256; i += 512) { h_s[i] = 0.f; hbf[i] = 0; }
  for (int i = tid; i < 32 * 64; i += 512) pbf[i] = 0;
  __syncthreads();

  for (int t = 0; t < 512; ++t) {
    if (t + 1 < 512)
      __builtin_prefetch((const char*)(gix + ((size_t)(t + 1) * 32) * 768) + tid * 96, 0, 1);
    // acc slots: 0 = r (gh+gi summed), 1 = z (summed), 2 = gh_n, 3 = gi_n
    v8f acc[2][4];
#pragma unroll
    for (int pp = 0; pp < 2; ++pp)
#pragma unroll
      for (int g4 = 0; g4 < 4; ++g4) acc[pp][g4] = vzero();

    const unsigned short* ar0 = &hbf[ln * 256];
    const unsigned short* ar1 = &hbf[(16 + ln) * 256];
    for (int ks = 0; ks < 8; ++ks) {               // h @ Whh^T (K=256)
      const int k0 = ks * 32 + gl * 8;
      const v16bf av0 = ldA(ar0, k0);
      const v16bf av1 = ldA(ar1, k0);
#pragma unroll
      for (int g4 = 0; g4 < 3; ++g4) {
        const v16bf bv = ldB(whh + (size_t)(g4 * 256 + w * 16 + ln) * 256 + ks * 32 + gl * 8);
        acc[0][g4] = wmma_bf16(av0, bv, acc[0][g4]);
        acc[1][g4] = wmma_bf16(av1, bv, acc[1][g4]);
      }
    }
    {
      const unsigned short* pr0 = &pbf[ln * 64];
      const unsigned short* pr1 = &pbf[(16 + ln) * 64];
      for (int ks = 0; ks < 2; ++ks) {             // prev @ Wp^T (K=64)
        const int k0 = ks * 32 + gl * 8;
        const v16bf av0 = ldA(pr0, k0);
        const v16bf av1 = ldA(pr1, k0);
#pragma unroll
        for (int g4 = 0; g4 < 3; ++g4) {
          const v16bf bv = ldB(wp + (size_t)(g4 * 256 + w * 16 + ln) * 64 + ks * 32 + gl * 8);
          const int dst = (g4 == 2) ? 3 : g4;      // keep gi_n separate
          acc[0][dst] = wmma_bf16(av0, bv, acc[0][dst]);
          acc[1][dst] = wmma_bf16(av1, bv, acc[1][dst]);
        }
      }
    }
    __syncthreads();
#pragma unroll
    for (int pp = 0; pp < 2; ++pp) {
      const int j = w * 16 + ln;
      const float br = bih[j] + bhh[j];
      const float bz = bih[256 + j] + bhh[256 + j];
      const float bin = bih[512 + j];
      const float bhn = bhh[512 + j];
#pragma unroll
      for (int r = 0; r < 8; ++r) {
        const int Mrow = pp * 16 + gl * 8 + r;
        const unsigned short* gp = gix + ((size_t)t * 32 + Mrow) * 768 + j;
        const float rg = sigm(acc[pp][0][r] + bf2f(gp[0]) + br);
        const float zg = sigm(acc[pp][1][r] + bf2f(gp[256]) + bz);
        const float hn = acc[pp][2][r] + bhn;
        const float nn = acc[pp][3][r] + bf2f(gp[512]) + bin;
        const float cand = tanhf(nn + rg * hn);
        const float ho = h_s[Mrow * 256 + j];
        const float hv = (1.f - zg) * cand + zg * ho;
        h_s[Mrow * 256 + j] = hv;
        hbf[Mrow * 256 + j] = f2bf(hv);
      }
    }
    __syncthreads();
    if (w < 8) {                                    // proj: h @ projW^T (32x64,K=256)
      const int mt = w & 1, ot = w >> 1;
      v8f pa = vzero();
      const unsigned short* ar = &hbf[(mt * 16 + ln) * 256];
      for (int ks = 0; ks < 8; ++ks) {
        const v16bf av = ldA(ar, ks * 32 + gl * 8);
        const v16bf bv = ldB(pw + (size_t)(ot * 16 + ln) * 256 + ks * 32 + gl * 8);
        pa = wmma_bf16(av, bv, pa);
      }
      const int o = ot * 16 + ln;
      const float bo = pb[o];
#pragma unroll
      for (int r = 0; r < 8; ++r) {
        const int Mrow = mt * 16 + gl * 8 + r;
        const float v = pa[r] + bo;
        pbf[Mrow * 64 + o] = f2bf(v);
        outs[(((size_t)Mrow * 512) + t) * 64 + o] = v;
      }
    }
    __syncthreads();
  }
}

// ---------------------------------------------------------------------------

extern "C" void kernel_launch(void* const* d_in, const int* in_sizes, int n_in,
                              void* d_out, int out_size, void* d_ws, size_t ws_size,
                              hipStream_t stream) {
  (void)in_sizes; (void)n_in; (void)out_size; (void)ws_size;
  const int*   enc_input = (const int*)  d_in[0];
  const float* spkr      = (const float*)d_in[2];
  const float* style     = (const float*)d_in[3];
  const float* speed     = (const float*)d_in[4];
  const float* emb       = (const float*)d_in[5];
  const float* conv_w    = (const float*)d_in[6];
  const float* conv_b    = (const float*)d_in[7];
  const float* bn_g      = (const float*)d_in[8];
  const float* bn_b      = (const float*)d_in[9];
  const float* bn_m      = (const float*)d_in[10];
  const float* bn_v      = (const float*)d_in[11];
  const float* lstm_wih  = (const float*)d_in[12];
  const float* lstm_whh  = (const float*)d_in[13];
  const float* lstm_bih  = (const float*)d_in[14];
  const float* lstm_bhh  = (const float*)d_in[15];
  const float* sp_w1     = (const float*)d_in[16];
  const float* sp_b1     = (const float*)d_in[17];
  const float* sp_w2     = (const float*)d_in[18];
  const float* sp_b2     = (const float*)d_in[19];
  const float* gru_wih   = (const float*)d_in[20];
  const float* gru_whh   = (const float*)d_in[21];
  const float* gru_bih   = (const float*)d_in[22];
  const float* gru_bhh   = (const float*)d_in[23];
  const float* proj_w    = (const float*)d_in[24];
  const float* proj_b    = (const float*)d_in[25];

  float* text = (float*)d_out;                       // (32,512,512)
  float* styp = text + (size_t)32 * 512 * 512;       // (32,512,64)

  char* ws = (char*)d_ws;
  size_t off = 0;
  auto carve = [&](size_t bytes) -> char* {
    char* p = ws + off; off += (bytes + 255) & ~(size_t)255; return p;
  };
  unsigned short* act_a = (unsigned short*)carve((size_t)32 * 516 * 512 * 2);
  unsigned short* act_b = (unsigned short*)carve((size_t)32 * 516 * 512 * 2);
  unsigned short* wkb   = (unsigned short*)carve((size_t)3 * 5 * 512 * 512 * 2);
  unsigned short* wlb   = (unsigned short*)carve((size_t)2 * 1024 * 512 * 2);
  unsigned short* whhb  = (unsigned short*)carve((size_t)2 * 1024 * 256 * 2);
  unsigned short* wgbx  = (unsigned short*)carve((size_t)768 * 512 * 2);
  unsigned short* wgbp  = (unsigned short*)carve((size_t)768 * 64 * 2);
  unsigned short* gwhhb = (unsigned short*)carve((size_t)768 * 256 * 2);
  unsigned short* pwb   = (unsigned short*)carve((size_t)64 * 256 * 2);
  unsigned short* gx    = (unsigned short*)carve((size_t)2 * 512 * 32 * 1024 * 2);
  unsigned short* gix   = (unsigned short*)carve((size_t)512 * 32 * 768 * 2);
  float*          spb   = (float*)carve((size_t)32 * 512 * 4);

  // Stage 0: embedding + weight prepacks (f32 -> bf16)
  embed_kernel<<<32 * 516, 256, 0, stream>>>(enc_input, emb, spkr, act_a);
  padzero_kernel<<<32 * 4, 256, 0, stream>>>(act_b);
  convw_prepack<<<(3 * 5 * 512 * 512 + 255) / 256, 256, 0, stream>>>(conv_w, wkb);
  cast_bf16<<<(2048 * 512 + 255) / 256, 256, 0, stream>>>(lstm_wih, wlb, 2048, 512, 0, 512);
  cast_bf16<<<(2048 * 256 + 255) / 256, 256, 0, stream>>>(lstm_whh, whhb, 2048, 256, 0, 256);
  cast_bf16<<<(768 * 512 + 255) / 256, 256, 0, stream>>>(gru_wih, wgbx, 768, 576, 0, 512);
  cast_bf16<<<(768 * 64 + 255) / 256, 256, 0, stream>>>(gru_wih, wgbp, 768, 576, 512, 64);
  cast_bf16<<<(768 * 256 + 255) / 256, 256, 0, stream>>>(gru_whh, gwhhb, 768, 256, 0, 256);
  cast_bf16<<<(64 * 256 + 255) / 256, 256, 0, stream>>>(proj_w, pwb, 64, 256, 0, 256);

  // Stage 1: 3 conv layers (WMMA GEMM + fused BN/ReLU/mask)
  dim3 cg(128, 8);
  conv_gemm_kernel<<<cg, 128, 0, stream>>>(act_a, wkb,                 conv_b,        bn_m,        bn_g,        bn_v,        bn_b,        spkr, act_b);
  conv_gemm_kernel<<<cg, 128, 0, stream>>>(act_b, wkb + 5 * 512 * 512, conv_b + 512,  bn_m + 512,  bn_g + 512,  bn_v + 512,  bn_b + 512,  spkr, act_a);
  conv_gemm_kernel<<<cg, 128, 0, stream>>>(act_a, wkb + 10 * 512 * 512, conv_b + 1024, bn_m + 1024, bn_g + 1024, bn_v + 1024, bn_b + 1024, spkr, act_b);

  // Stage 2: LSTM input-gate precompute (both directions) then recurrences
  dim3 gg(128, 16);
  gemm_bf16_kernel<<<gg, 128, 0, stream>>>(act_b, 516, 2, wlb, gx, 1024);
  gemm_bf16_kernel<<<gg, 128, 0, stream>>>(act_b, 516, 2, wlb + 1024 * 512,
                                           gx + (size_t)512 * 32 * 1024, 1024);
  lstm_kernel<<<1, 512, 0, stream>>>(gx, whhb, lstm_bih, lstm_bhh, text, 0, 0);
  lstm_kernel<<<1, 512, 0, stream>>>(gx + (size_t)512 * 32 * 1024, whhb + 1024 * 256,
                                     lstm_bih + 1024, lstm_bhh + 1024, text, 1, 256);

  // Stage 3: text_enc += style + speed-MLP; make bf16 copy for GRU GEMM
  sp_kernel<<<64, 256, 0, stream>>>(speed, sp_w1, sp_b1, sp_w2, sp_b2, spb);
  add_style_kernel<<<32 * 512, 256, 0, stream>>>(text, style, spb, act_a);

  // Stage 4: GRU input-gate precompute then recurrence + projection
  dim3 gg2(128, 12);
  gemm_bf16_kernel<<<gg2, 128, 0, stream>>>(act_a, 512, 0, wgbx, gix, 768);
  gru_kernel<<<1, 512, 0, stream>>>(gix, gwhhb, wgbp, pwb, gru_bih, gru_bhh, proj_b, styp);
}
